// QuanvolutionAutoencoder_65481071395629
// MI455X (gfx1250) — compile-verified, matrix operand order checked by reading, not run on hardware
//
#include <hip/hip_runtime.h>
#include <hip/hip_bf16.h>

typedef __attribute__((ext_vector_type(16))) _Float16 v16h;
typedef __attribute__((ext_vector_type(8)))  _Float16 v8h;
typedef __attribute__((ext_vector_type(8)))  float    v8f;

#define BQ 4096

// ---------------------------------------------------------------------------
// Quanvolution: one thread per 2x2 patch. 4-qubit (16-amplitude) real
// statevector sim, 8 fixed CRY gates, Pauli-Z expectation per wire.
// Writes f16 features into K-padded [4096][800] activation buffer.
// ---------------------------------------------------------------------------
__global__ void quanv_kernel(const float* __restrict__ x,
                             const float* __restrict__ theta,
                             _Float16* __restrict__ feats) {
  int idx = blockIdx.x * blockDim.x + threadIdx.x;
  if (idx >= BQ * 196) return;
  int b  = idx / 196;
  int p  = idx - b * 196;
  int pr = p / 14;
  int pc = p - pr * 14;
  const float* img = x + (size_t)b * 784;
  int r0 = 2 * pr, c0 = 2 * pc;
  float ang[4];
  ang[0] = img[r0 * 28 + c0];
  ang[1] = img[r0 * 28 + c0 + 1];
  ang[2] = img[(r0 + 1) * 28 + c0];
  ang[3] = img[(r0 + 1) * 28 + c0 + 1];
  float cw[4], sw[4];
#pragma unroll
  for (int w = 0; w < 4; ++w) sincosf(0.5f * ang[w], &sw[w], &cw[w]);

  // product state: st[i], wire w is bit (3-w) of i
  float st[16];
#pragma unroll
  for (int i = 0; i < 16; ++i) {
    float v = ((i >> 3) & 1) ? sw[0] : cw[0];
    v *= ((i >> 2) & 1) ? sw[1] : cw[1];
    v *= ((i >> 1) & 1) ? sw[2] : cw[2];
    v *= (i & 1) ? sw[3] : cw[3];
    st[i] = v;
  }

  constexpr int ctrlw[8] = {0, 1, 2, 3, 0, 1, 2, 3};
  constexpr int tgtw[8]  = {1, 2, 3, 0, 2, 3, 0, 1};
#pragma unroll
  for (int k = 0; k < 8; ++k) {
    float gs, gc;
    sincosf(0.5f * theta[k], &gs, &gc);
    const int cb = 3 - ctrlw[k], tb = 3 - tgtw[k];
#pragma unroll
    for (int i = 0; i < 16; ++i) {
      if (((i >> cb) & 1) && !((i >> tb) & 1)) {
        int j = i | (1 << tb);
        float v0 = st[i], v1 = st[j];
        st[i] = gc * v0 - gs * v1;
        st[j] = gs * v0 + gc * v1;
      }
    }
  }

  float m0 = 0.f, m1 = 0.f, m2 = 0.f, m3 = 0.f;
#pragma unroll
  for (int i = 0; i < 16; ++i) {
    float p2 = st[i] * st[i];
    m0 += ((i >> 3) & 1) ? -p2 : p2;
    m1 += ((i >> 2) & 1) ? -p2 : p2;
    m2 += ((i >> 1) & 1) ? -p2 : p2;
    m3 += (i & 1) ? -p2 : p2;
  }
  _Float16* o = feats + (size_t)b * 800 + p * 4;
  o[0] = (_Float16)m0;
  o[1] = (_Float16)m1;
  o[2] = (_Float16)m2;
  o[3] = (_Float16)m3;
}

// Zero the K padding columns [784..799] of the feats buffer.
__global__ void pad_feats_kernel(_Float16* __restrict__ feats) {
  int i = blockIdx.x * blockDim.x + threadIdx.x;
  if (i >= BQ * 16) return;
  int b = i >> 4;
  feats[(size_t)b * 800 + 784 + (i & 15)] = (_Float16)0.0f;
}

// Convert W[K][N] fp32 -> Wt[Npad][Kpad] f16 (transpose + zero pad both dims).
__global__ void transpose_to_f16(const float* __restrict__ W,
                                 _Float16* __restrict__ Wt,
                                 int K, int N, int Kpad, int Npad) {
  int i = blockIdx.x * blockDim.x + threadIdx.x;
  if (i >= Npad * Kpad) return;
  int n  = i / Kpad;
  int kp = i - n * Kpad;
  Wt[(size_t)n * Kpad + kp] =
      (kp < K && n < N) ? (_Float16)W[(size_t)kp * N + n] : (_Float16)0.0f;
}

// Per-lane WMMA 16x32 f16 fragment load (ISA 16-bit A-matrix layout):
//   lane<16 : row=lane,    halves K{0..7} then K{16..23}
//   lane>=16: row=lane-16, halves K{8..15} then K{24..31}
// Caller supplies p = base + row*K + koff, with koff = (lane>>4)*8.
__device__ __forceinline__ v16h load_frag(const _Float16* __restrict__ p) {
  v8h lo = *(const v8h*)p;
  v8h hi = *(const v8h*)(p + 16);
  return __builtin_shufflevector(lo, hi, 0, 1, 2, 3, 4, 5, 6, 7,
                                 8, 9, 10, 11, 12, 13, 14, 15);
}

__device__ __forceinline__ v8f wmma_f16(v16h a, v16h b, v8f c) {
  return __builtin_amdgcn_wmma_f32_16x16x32_f16(
      /*neg_a=*/false, a, /*neg_b=*/false, b,
      /*c_mod=*/(short)0, c, /*reuse_a=*/false, /*reuse_b=*/false);
}

// ---------------------------------------------------------------------------
// GEMM: C[M,N] = act(A[M,K] @ B[K,N] + bias), B pre-transposed/padded as
// Bt[Npad][K] f16. One wave per 32x32 C tile: 2x2 register blocking ->
// 4 accumulators, 4 WMMAs per K-step from 2 A + 2 B fragments (2x reuse).
// Stores (and bias reads) are guarded by ocol<N; all WMMAs run with EXEC
// all-ones (guards only in the epilogue, early-exit is wave-uniform).
// ACT: 0 = relu -> f16 out, 1 = identity -> f16 out, 2 = sigmoid -> f32 out
// ---------------------------------------------------------------------------
template <int ACT>
__global__ void gemm_wmma_kernel(const _Float16* __restrict__ A,
                                 const _Float16* __restrict__ Bt,
                                 const float* __restrict__ bias,
                                 void* __restrict__ Cout,
                                 int M, int N, int K, int ldc, int Npad) {
  const int wavesPerBlk = blockDim.x >> 5;
  int wave   = blockIdx.x * wavesPerBlk + (threadIdx.x >> 5);
  int tilesN = Npad >> 5;
  int tm = (wave / tilesN) << 5;
  int tn = (wave % tilesN) << 5;
  if (tm >= M) return;  // wave-uniform: EXEC stays all-ones for WMMA

  int lane = threadIdx.x & 31;
  int r    = lane & 15;
  int koff = (lane >> 4) << 3;
  const _Float16* ap0 = A + (size_t)(tm + r) * K + koff;
  const _Float16* ap1 = ap0 + (size_t)16 * K;
  const _Float16* bp0 = Bt + (size_t)(tn + r) * K + koff;
  const _Float16* bp1 = bp0 + (size_t)16 * K;

  v8f acc[2][2] = {{{}, {}}, {{}, {}}};
#pragma unroll 2
  for (int k = 0; k < K; k += 32) {
    v16h a0 = load_frag(ap0 + k);
    v16h a1 = load_frag(ap1 + k);
    v16h b0 = load_frag(bp0 + k);
    v16h b1 = load_frag(bp1 + k);
    acc[0][0] = wmma_f16(a0, b0, acc[0][0]);
    acc[0][1] = wmma_f16(a0, b1, acc[0][1]);
    acc[1][0] = wmma_f16(a1, b0, acc[1][0]);
    acc[1][1] = wmma_f16(a1, b1, acc[1][1]);
  }

  // C/D layout per 16x16 tile: VGPR rr -> row rr (lanes 0-15) / rr+8
  // (lanes 16-31), col = lane & 15.
  int rbase = (lane < 16) ? 0 : 8;
  int c     = lane & 15;
#pragma unroll
  for (int ni = 0; ni < 2; ++ni) {
    int ocol = tn + ni * 16 + c;
    if (ocol >= N) continue;  // epilogue-only divergence (N padding)
    float bb = bias[ocol];
#pragma unroll
    for (int mi = 0; mi < 2; ++mi) {
      int orow = tm + mi * 16 + rbase;
#pragma unroll
      for (int rr = 0; rr < 8; ++rr) {
        float v = acc[mi][ni][rr] + bb;
        if (ACT == 0) v = fmaxf(v, 0.0f);
        if (ACT == 2) {
          v = 1.0f / (1.0f + __expf(-v));
          ((float*)Cout)[(size_t)(orow + rr) * ldc + ocol] = v;
        } else {
          ((_Float16*)Cout)[(size_t)(orow + rr) * ldc + ocol] = (_Float16)v;
        }
      }
    }
  }
}

// ---------------------------------------------------------------------------
extern "C" void kernel_launch(void* const* d_in, const int* in_sizes, int n_in,
                              void* d_out, int out_size, void* d_ws, size_t ws_size,
                              hipStream_t stream) {
  (void)in_sizes; (void)n_in; (void)out_size; (void)ws_size;
  const float* x  = (const float*)d_in[0];
  const float* th = (const float*)d_in[1];
  const float* W1 = (const float*)d_in[2];
  const float* b1 = (const float*)d_in[3];
  const float* W2 = (const float*)d_in[4];
  const float* b2 = (const float*)d_in[5];
  const float* W3 = (const float*)d_in[6];
  const float* b3 = (const float*)d_in[7];
  const float* W4 = (const float*)d_in[8];
  const float* b4 = (const float*)d_in[9];
  float* out = (float*)d_out;

  char* ws = (char*)d_ws;
  size_t off = 0;
  auto take = [&](size_t bytes) -> char* {
    char* p = ws + off;
    off += (bytes + 255) & ~(size_t)255;
    return p;
  };
  _Float16* feats = (_Float16*)take((size_t)4096 * 800 * 2);  // A1 (Kpad=800)
  _Float16* W1t   = (_Float16*)take((size_t)512 * 800 * 2);   // Npad=512
  _Float16* h1    = (_Float16*)take((size_t)4096 * 512 * 2);
  _Float16* W2t   = (_Float16*)take((size_t)32 * 512 * 2);    // Npad=32
  _Float16* lat   = (_Float16*)take((size_t)4096 * 32 * 2);
  _Float16* W3t   = (_Float16*)take((size_t)512 * 32 * 2);    // Npad=512
  _Float16* h2    = (_Float16*)take((size_t)4096 * 512 * 2);
  _Float16* W4t   = (_Float16*)take((size_t)800 * 512 * 2);   // Npad=800 (N=784)

  // Weight conversion (transpose to [Npad][Kpad] f16, zero-padded)
  {
    int n = 512 * 800;
    transpose_to_f16<<<(n + 255) / 256, 256, 0, stream>>>(W1, W1t, 784, 512, 800, 512);
  }
  {
    int n = 32 * 512;
    transpose_to_f16<<<(n + 255) / 256, 256, 0, stream>>>(W2, W2t, 512, 32, 512, 32);
  }
  {
    int n = 512 * 32;
    transpose_to_f16<<<(n + 255) / 256, 256, 0, stream>>>(W3, W3t, 32, 512, 32, 512);
  }
  {
    int n = 800 * 512;
    transpose_to_f16<<<(n + 255) / 256, 256, 0, stream>>>(W4, W4t, 512, 784, 512, 800);
  }

  // Quanvolution features
  {
    int n = BQ * 196;
    quanv_kernel<<<(n + 127) / 128, 128, 0, stream>>>(x, th, feats);
  }
  {
    int n = BQ * 16;
    pad_feats_kernel<<<(n + 255) / 256, 256, 0, stream>>>(feats);
  }

  // MLP: 4 WMMA GEMMs, one wave per 32x32 tile, 4 waves / 128-thread block
  {
    int waves = (4096 >> 5) * (512 >> 5);  // 2048
    gemm_wmma_kernel<0><<<(waves + 3) / 4, 128, 0, stream>>>(
        feats, W1t, b1, h1, 4096, 512, 800, 512, 512);
  }
  {
    int waves = (4096 >> 5) * (32 >> 5);   // 128
    gemm_wmma_kernel<1><<<(waves + 3) / 4, 128, 0, stream>>>(
        h1, W2t, b2, lat, 4096, 32, 512, 32, 32);
  }
  {
    int waves = (4096 >> 5) * (512 >> 5);  // 2048
    gemm_wmma_kernel<0><<<(waves + 3) / 4, 128, 0, stream>>>(
        lat, W3t, b3, h2, 4096, 512, 32, 512, 512);
  }
  {
    int waves = (4096 >> 5) * (800 >> 5);  // 3200
    gemm_wmma_kernel<2><<<(waves + 3) / 4, 128, 0, stream>>>(
        h2, W4t, b4, out, 4096, 784, 512, 784, 800);
  }
}
